// EdgeScoreHead_25829933318905
// MI455X (gfx1250) — matrix operand order repaired and checked by reading, not compiled, not run on hardware
//
#include <hip/hip_runtime.h>

#define IN_DIM   64
#define EDGE_DIM 8
#define FEAT     136
#define FEATP    160     // padded to 5 K-chunks of 32 (k=136 carries bias via 1.0)
#define H1       128
#define H2       64
#define TM       2       // M-tiles (of 16 edges) per wave -> 32 edges/wave

#define W1_ELEMS (5 * 8 * 512)   // packed W1 bf16 elements
#define W2_ELEMS (4 * 4 * 512)   // packed W2 bf16 elements
#define WPACK_BYTES ((W1_ELEMS + W2_ELEMS) * 2)   // 57344, 256-aligned

typedef __attribute__((ext_vector_type(16))) __bf16 v16bf;
typedef __attribute__((ext_vector_type(8)))  float  v8f;

union Frag16 { uint4 q[2]; v16bf v; };

__device__ __forceinline__ unsigned short f2bf(float f) {
    unsigned int u = __float_as_uint(f);
    // round-to-nearest-even f32 -> bf16
    unsigned int r = (u + 0x7FFFu + ((u >> 16) & 1u)) >> 16;
    return (unsigned short)r;
}

__device__ __forceinline__ uint4 pack8(float4 a, float4 b) {
    uint4 p;
    p.x = (unsigned)f2bf(a.x) | ((unsigned)f2bf(a.y) << 16);
    p.y = (unsigned)f2bf(a.z) | ((unsigned)f2bf(a.w) << 16);
    p.z = (unsigned)f2bf(b.x) | ((unsigned)f2bf(b.y) << 16);
    p.w = (unsigned)f2bf(b.z) | ((unsigned)f2bf(b.w) << 16);
    return p;
}

// ---------------------------------------------------------------------------
// Pack W1 [FEAT x H1] into bf16 B-fragment layout; bias b1 folded in as the
// K=136 row (feat rows carry 1.0 there).  Fragment f = kc*8 + nt; lane L
// holds 16 consecutive bf16 for column n = nt*16 + L%16,
// K = kc*32 + (j<8 ? 0 : 16) + (L<16 ? 0 : 8) + (j & 7).
// ---------------------------------------------------------------------------
__global__ void pack_w1(const float* __restrict__ W1, const float* __restrict__ b1,
                        unsigned short* __restrict__ out) {
    int o = blockIdx.x * blockDim.x + threadIdx.x;
    if (o >= W1_ELEMS) return;
    int frag   = o >> 9;
    int within = o & 511;
    int lane   = within >> 4;
    int j      = within & 15;
    int kc = frag >> 3;
    int nt = frag & 7;
    int n  = nt * 16 + (lane & 15);
    int k  = kc * 32 + ((j < 8) ? 0 : 16) + ((lane < 16) ? 0 : 8) + (j & 7);
    float v = (k < FEAT) ? W1[k * H1 + n] : ((k == FEAT) ? b1[n] : 0.0f);
    out[o] = f2bf(v);
}

// W2 [H1 x H2]: kc in 0..3, nt in 0..3, no padding needed.
__global__ void pack_w2(const float* __restrict__ W2, unsigned short* __restrict__ out) {
    int o = blockIdx.x * blockDim.x + threadIdx.x;
    if (o >= W2_ELEMS) return;
    int frag   = o >> 9;
    int within = o & 511;
    int lane   = within >> 4;
    int j      = within & 15;
    int kc = frag >> 2;
    int nt = frag & 3;
    int n  = nt * 16 + (lane & 15);
    int k  = kc * 32 + ((j < 8) ? 0 : 16) + ((lane < 16) ? 0 : 8) + (j & 7);
    out[o] = f2bf(W2[k * H2 + n]);
}

// Pre-convert node embeddings z (f32) to bf16: halves gather traffic and
// removes all conversion VALU from the hot kernel.  8 elements per thread.
__global__ void conv_z(const float* __restrict__ zin, unsigned short* __restrict__ zout,
                       int n8) {
    int o = blockIdx.x * blockDim.x + threadIdx.x;
    if (o >= n8) return;
    const float4* p = (const float4*)(zin + (long long)o * 8);
    ((uint4*)zout)[o] = pack8(p[0], p[1]);
}

// ---------------------------------------------------------------------------
// Fused kernel: 1 wave = 32 edges (2 M-tiles of 16 sharing every B fragment).
//   Layer 1: 8 nt x 5 kc x 2 tiles = 80 WMMAs (bias folded into K=136 row)
//   Layer 2: 4 nt x 4 kc x 2 tiles = 32 WMMAs
//   Layer 3: VALU dot in C/D layout + shfl_xor butterfly per tile
// PRECONV: gather from bf16 z (pure b128 copies); else convert f32 on the fly.
// ---------------------------------------------------------------------------
template <bool PRECONV>
__global__ __launch_bounds__(128)
void edge_mlp(const float* __restrict__ z,
              const unsigned short* __restrict__ zb,
              const long long* __restrict__ eidx,
              const float* __restrict__ eattr,
              const unsigned short* __restrict__ w1p,
              const unsigned short* __restrict__ w2p,
              const float* __restrict__ b2,
              const float* __restrict__ W3,
              const float* __restrict__ b3,
              float* __restrict__ out, int E)
{
    __shared__ unsigned short sBuf[4][32][FEATP];   // 40960 B

    const int wave = threadIdx.x >> 5;
    const int lane = threadIdx.x & 31;
    const int hi   = (lane >= 16) ? 1 : 0;
    const int ln   = lane & 15;
    const long long ebase = (long long)blockIdx.x * 128 + (long long)wave * 32;

    // ---- Phase A: gather feat rows into LDS; 1 lane = 1 edge ---------------
    {
        long long ge  = ebase + lane;
        long long gec = (ge < (long long)E) ? ge : (long long)(E - 1);
        int s = (int)eidx[gec];
        int d = (int)eidx[(long long)E + gec];

        uint4* row = (uint4*)&sBuf[wave][lane][0];    // 20 uint4 = 160 bf16
        if (PRECONV) {
            const uint4* zs = (const uint4*)(zb + (long long)s * IN_DIM);
            const uint4* zd = (const uint4*)(zb + (long long)d * IN_DIM);
            #pragma unroll
            for (int q = 0; q < 8; ++q) row[q]     = zs[q];
            #pragma unroll
            for (int q = 0; q < 8; ++q) row[8 + q] = zd[q];
        } else {
            const float4* zs = (const float4*)(z + (long long)s * IN_DIM);
            const float4* zd = (const float4*)(z + (long long)d * IN_DIM);
            #pragma unroll
            for (int q = 0; q < 8; ++q) row[q]     = pack8(zs[2 * q], zs[2 * q + 1]);
            #pragma unroll
            for (int q = 0; q < 8; ++q) row[8 + q] = pack8(zd[2 * q], zd[2 * q + 1]);
        }
        const float4* ea = (const float4*)(eattr + gec * EDGE_DIM);
        row[16] = pack8(ea[0], ea[1]);                // bf16 128..135
        uint4 onep = {0x00003F80u, 0u, 0u, 0u};       // bf16 1.0 at k=136 (bias)
        uint4 zero = {0u, 0u, 0u, 0u};
        row[17] = onep; row[18] = zero; row[19] = zero;
    }
    __syncthreads();

    // ---- Layer-1 A fragments for both M-tiles into registers ---------------
    Frag16 aF[TM][5];
    #pragma unroll
    for (int t = 0; t < TM; ++t) {
        #pragma unroll
        for (int kc = 0; kc < 5; ++kc) {
            aF[t][kc].q[0] = *(const uint4*)&sBuf[wave][t * 16 + ln][kc * 32 + hi * 8];
            aF[t][kc].q[1] = *(const uint4*)&sBuf[wave][t * 16 + ln][kc * 32 + 16 + hi * 8];
        }
    }
    // feat region now dead for this wave; h1 overlays it.

    // ---- Layer 1: each B fragment feeds both M-tiles; C starts at 0 --------
    #pragma unroll
    for (int nt = 0; nt < 8; ++nt) {
        v8f c0 = {}, c1 = {};
        #pragma unroll
        for (int kc = 0; kc < 5; ++kc) {
            Frag16 bF;
            const uint4* bp = (const uint4*)(w1p + ((kc * 8 + nt) * 512 + lane * 16));
            bF.q[0] = bp[0];
            bF.q[1] = bp[1];
            c0 = __builtin_amdgcn_wmma_f32_16x16x32_bf16(
                     false, aF[0][kc].v, false, bF.v, (short)0, c0, false, false);
            c1 = __builtin_amdgcn_wmma_f32_16x16x32_bf16(
                     false, aF[1][kc].v, false, bF.v, (short)0, c1, false, false);
        }
        // ReLU (v_max) + RNE bf16, scatter into h1 overlay
        #pragma unroll
        for (int r = 0; r < 8; ++r) {
            float v0 = fmaxf(c0[r], 0.0f);
            float v1 = fmaxf(c1[r], 0.0f);
            sBuf[wave][r + hi * 8][nt * 16 + ln]      = f2bf(v0);
            sBuf[wave][16 + r + hi * 8][nt * 16 + ln] = f2bf(v1);
        }
    }
    __syncthreads();

    // ---- Layer 2: A from h1 overlay, each B fragment feeds both tiles ------
    Frag16 a2[TM][4];
    #pragma unroll
    for (int t = 0; t < TM; ++t) {
        #pragma unroll
        for (int kc = 0; kc < 4; ++kc) {
            a2[t][kc].q[0] = *(const uint4*)&sBuf[wave][t * 16 + ln][kc * 32 + hi * 8];
            a2[t][kc].q[1] = *(const uint4*)&sBuf[wave][t * 16 + ln][kc * 32 + 16 + hi * 8];
        }
    }

    v8f h2acc[TM][4];
    #pragma unroll
    for (int nt = 0; nt < 4; ++nt) {
        float bias = b2[nt * 16 + ln];
        v8f c0, c1;
        #pragma unroll
        for (int r = 0; r < 8; ++r) { c0[r] = bias; c1[r] = bias; }
        #pragma unroll
        for (int kc = 0; kc < 4; ++kc) {
            Frag16 bF;
            const uint4* bp = (const uint4*)(w2p + ((kc * 4 + nt) * 512 + lane * 16));
            bF.q[0] = bp[0];
            bF.q[1] = bp[1];
            c0 = __builtin_amdgcn_wmma_f32_16x16x32_bf16(
                     false, a2[0][kc].v, false, bF.v, (short)0, c0, false, false);
            c1 = __builtin_amdgcn_wmma_f32_16x16x32_bf16(
                     false, a2[1][kc].v, false, bF.v, (short)0, c1, false, false);
        }
        #pragma unroll
        for (int r = 0; r < 8; ++r) {
            h2acc[0][nt][r] = fmaxf(c0[r], 0.0f);
            h2acc[1][nt][r] = fmaxf(c1[r], 0.0f);
        }
    }

    // ---- Layer 3 (64 -> 1) in C/D layout, per M-tile -----------------------
    float w3l[4];
    #pragma unroll
    for (int nt = 0; nt < 4; ++nt) w3l[nt] = W3[nt * 16 + ln];
    float bb = b3[0];

    #pragma unroll
    for (int t = 0; t < TM; ++t) {
        float p[8];
        #pragma unroll
        for (int r = 0; r < 8; ++r)
            p[r] = h2acc[t][0][r] * w3l[0] + h2acc[t][1][r] * w3l[1]
                 + h2acc[t][2][r] * w3l[2] + h2acc[t][3][r] * w3l[3];

        #pragma unroll
        for (int m = 1; m < 16; m <<= 1) {
            #pragma unroll
            for (int r = 0; r < 8; ++r) p[r] += __shfl_xor(p[r], m, 32);
        }

        if (ln == 0) {  // lane 0 -> rows 0..7, lane 16 -> rows 8..15
            long long rowbase = ebase + t * 16 + hi * 8;
            #pragma unroll
            for (int r = 0; r < 8; ++r) {
                long long ge = rowbase + r;
                if (ge < (long long)E) out[ge] = p[r] + bb;
            }
        }
    }
}

extern "C" void kernel_launch(void* const* d_in, const int* in_sizes, int n_in,
                              void* d_out, int out_size, void* d_ws, size_t ws_size,
                              hipStream_t stream) {
    const float*     z     = (const float*)d_in[0];
    const long long* eidx  = (const long long*)d_in[1];   // int64 edge_index [2, E]
    const float*     eattr = (const float*)d_in[2];
    const float*     W1    = (const float*)d_in[3];
    const float*     b1    = (const float*)d_in[4];
    const float*     W2    = (const float*)d_in[5];
    const float*     b2    = (const float*)d_in[6];
    const float*     W3    = (const float*)d_in[7];
    const float*     b3    = (const float*)d_in[8];
    float*           out   = (float*)d_out;

    int E  = in_sizes[1] / 2;     // edge_index has 2*E int64 elements
    int ZN = in_sizes[0];         // N_NODES * IN_DIM f32 elements

    unsigned short* w1p = (unsigned short*)d_ws;
    unsigned short* w2p = w1p + W1_ELEMS;
    unsigned short* zb  = (unsigned short*)((char*)d_ws + WPACK_BYTES);

    pack_w1<<<(W1_ELEMS + 255) / 256, 256, 0, stream>>>(W1, b1, w1p);
    pack_w2<<<(W2_ELEMS + 255) / 256, 256, 0, stream>>>(W2, w2p);

    bool preconv = ws_size >= (size_t)WPACK_BYTES + (size_t)ZN * 2;
    if (preconv) {
        int n8 = ZN / 8;
        conv_z<<<(n8 + 255) / 256, 256, 0, stream>>>(z, zb, n8);
    }

    int blocks = (E + 127) / 128;   // 128 edges per 128-thread block (4 waves)
    if (preconv) {
        edge_mlp<true><<<blocks, 128, 0, stream>>>(z, zb, eidx, eattr, w1p, w2p,
                                                   b2, W3, b3, out, E);
    } else {
        edge_mlp<false><<<blocks, 128, 0, stream>>>(z, zb, eidx, eattr, w1p, w2p,
                                                    b2, W3, b3, out, E);
    }
}
